// FeatureTokenizer_9818295239297
// MI455X (gfx1250) — matrix operand order repaired and checked by reading, not compiled, not run on hardware
//
#include <hip/hip_runtime.h>
#include <hip/hip_bf16.h>
#include <math.h>

#define BATCH 256
#define NFEAT 10000
#define HID   768
#define KTOK  512

#if defined(__has_builtin)
#if __has_builtin(__builtin_amdgcn_sched_barrier)
#define SCHED_FENCE() __builtin_amdgcn_sched_barrier(0)
#endif
#endif
#ifndef SCHED_FENCE
#define SCHED_FENCE()
#endif

typedef __bf16 v16bf __attribute__((ext_vector_type(16)));
typedef float  v8f   __attribute__((ext_vector_type(8)));

__device__ __forceinline__ unsigned short f2bf(float f) {
  unsigned u = __float_as_uint(f);
  u += 0x7FFFu + ((u >> 16) & 1u);           // round-to-nearest-even
  return (unsigned short)(u >> 16);
}

// ---------------------------------------------------------------------------
// Kernel 1: repack w2 [K=768, N=768] (row-major f32) into bf16 WMMA B-fragment
// layout: B_pack[ntile=48][kstep=24][lane=32][16 bf16], where for a 32x16 KxN
// fragment: lane<16 holds column N=lane, K=kbase+0..15; lane>=16 holds column
// N=lane-16, K=kbase+16..31 (element j <-> K offset j within its half).
// ---------------------------------------------------------------------------
__global__ __launch_bounds__(256) void pack_w2_kernel(
    const float* __restrict__ w2, unsigned short* __restrict__ bp) {
  int e = blockIdx.x * 256 + threadIdx.x;
  if (e >= HID * HID) return;
  int h = e / HID;            // K row
  int d = e - h * HID;        // N col
  int nt = d >> 4;
  int ks = h >> 5;
  int kk = h & 31;
  int L  = (d & 15) + ((kk & 16) ? 16 : 0);
  int j  = kk & 15;
  bp[(((nt * 24 + ks) * 32) + L) * 16 + j] = f2bf(w2[e]);
}

// ---------------------------------------------------------------------------
// Kernel 2: per-row exact top-512 of |x| (sorted descending), one block/row.
// Radix-select (4x 8-bit MSB-first passes) finds the threshold, gather fills
// exactly 512 candidates, bitonic sort orders them. Emits feature ids, signed
// values, attention mask and empty-row flags.
// ---------------------------------------------------------------------------
__global__ __launch_bounds__(256) void topk_kernel(
    const float* __restrict__ x, int* __restrict__ fid,
    float* __restrict__ vals, int* __restrict__ mask,
    int* __restrict__ empty) {
  __shared__ float xrow[NFEAT];                       // 40 KB
  __shared__ unsigned int hist[256];
  __shared__ unsigned long long cand[KTOK];           // key<<32 | idx
  __shared__ unsigned int s_thresh, s_need, s_cntg, s_cnte, s_nact;
  const int b = blockIdx.x, tid = threadIdx.x;
  const float* xr = x + (size_t)b * NFEAT;
  for (int i = tid; i < NFEAT; i += 256) xrow[i] = xr[i];
  if (tid == 0) { s_need = KTOK; s_thresh = 0u; s_cntg = 0u; s_cnte = 0u; s_nact = 0u; }
  __syncthreads();

  for (int shift = 24; shift >= 0; shift -= 8) {
    hist[tid] = 0u;
    __syncthreads();
    unsigned hib  = s_thresh;
    unsigned need = s_need;
    for (int i = tid; i < NFEAT; i += 256) {
      unsigned u = __float_as_uint(fabsf(xrow[i]));
      bool match = (shift == 24) || ((u >> (shift + 8)) == hib);
      if (match) atomicAdd(&hist[(u >> shift) & 255u], 1u);
    }
    __syncthreads();
    if (tid == 0) {
      unsigned c = 0; int chosen = 0; unsigned cg = 0;
      for (int bin = 255; bin >= 0; --bin) {
        unsigned c2 = c + hist[bin];
        if (c2 >= need) { chosen = bin; cg = c; break; }
        c = c2;
      }
      s_thresh = (hib << 8) | (unsigned)chosen;
      s_need   = need - cg;                  // how many ==threshold to keep
    }
    __syncthreads();
  }
  const unsigned T = s_thresh;
  const unsigned need = s_need;

  // gather: strictly greater first, then exactly `need` equal-to-threshold
  for (int i = tid; i < NFEAT; i += 256) {
    unsigned u = __float_as_uint(fabsf(xrow[i]));
    if (u > T) {
      unsigned p = atomicAdd(&s_cntg, 1u);
      cand[p] = ((unsigned long long)u << 32) | (unsigned)i;
    }
  }
  __syncthreads();
  const unsigned G = s_cntg;
  for (int i = tid; i < NFEAT; i += 256) {
    unsigned u = __float_as_uint(fabsf(xrow[i]));
    if (u == T) {
      unsigned q = atomicAdd(&s_cnte, 1u);
      if (q < need) cand[G + q] = ((unsigned long long)u << 32) | (unsigned)i;
    }
  }
  __syncthreads();

  // bitonic sort, descending (512 elems, 256 threads = 1 pair/thread/pass)
  for (int size = 2; size <= KTOK; size <<= 1) {
    for (int j = size >> 1; j > 0; j >>= 1) {
      int a  = ((tid / j) * 2 * j) + (tid % j);
      int bb = a + j;
      bool up = ((a & size) == 0);
      unsigned long long va = cand[a], vb = cand[bb];
      if (up ? (va < vb) : (va > vb)) { cand[a] = vb; cand[bb] = va; }
      __syncthreads();
    }
  }

  unsigned nact = 0;
  for (int s = tid; s < KTOK; s += 256) {
    unsigned long long e = cand[s];
    unsigned u = (unsigned)(e >> 32);
    int idx = (int)(e & 0xFFFFFFFFull);
    int act = (u > 0u) ? 1 : 0;
    fid[b * KTOK + s]  = act ? (idx + 1) : 0;
    vals[b * KTOK + s] = act ? xrow[idx] : 0.0f;
    mask[b * KTOK + s] = act;
    nact += (unsigned)act;
  }
  atomicAdd(&s_nact, nact);
  __syncthreads();
  if (tid == 0) {
    int em = (s_nact == 0u) ? 1 : 0;
    empty[b] = em;
    if (em) mask[b * KTOK + 0] = 1;
  }
}

// ---------------------------------------------------------------------------
// Kernel 3: fused GELU-projection GEMM (bf16 WMMA) + emb gather + LayerNorm.
// One block = 16 tokens x full H. 8 waves; each wave owns 6 N-tiles of 16.
// A tile (16x768, GELU(v*w1+b1) -> bf16 fragment layout) staged once in LDS.
// K loop is software-pipelined with prefetch distance 1; sched_barrier fences
// stop the machine scheduler from sinking the t+1 prefetch loads below the
// step-t WMMAs, so every WMMA's operands were loaded a full iteration earlier.
// ---------------------------------------------------------------------------
__global__ __launch_bounds__(256) void fused_kernel(
    const float* __restrict__ emb, const float* __restrict__ w1,
    const float* __restrict__ b1,  const float* __restrict__ b2,
    const float* __restrict__ lnw, const float* __restrict__ lnb,
    const unsigned short* __restrict__ bp,
    const int* __restrict__ fid, const float* __restrict__ vals,
    const int* __restrict__ empty, float* __restrict__ out) {
  __shared__ __align__(32) char smem[16 * HID * 4];   // 48 KB, dual-purpose
  unsigned short* A_lds = (unsigned short*)smem;      // 24 KB live in phase 1
  float* h_lds = (float*)smem;                        // 48 KB live in phase 2
  __shared__ float vals_s[16];
  __shared__ int   fid_s[16];
  __shared__ int   empty_s;

  const int tid  = threadIdx.x;
  const int b    = blockIdx.x >> 5;      // 32 tiles of 16 tokens per row
  const int tile = blockIdx.x & 31;
  const int tokbase = b * KTOK + tile * 16;

  if (tid < 16) { vals_s[tid] = vals[tokbase + tid]; fid_s[tid] = fid[tokbase + tid]; }
  if (tid == 0) empty_s = (tile == 0) ? empty[b] : 0;
  __syncthreads();

  // ---- build A fragments: A[m,k] = bf16(GELU(v[m]*w1[k]+b1[k])) ----
  // A 16x32 layout used here (contiguous k, same as B_pack): lane<16 -> M=lane, Koff 0..15;
  //                        lane>=16 -> M=lane-16, Koff 16..31
  for (int e = tid; e < 16 * HID; e += 256) {
    int m = e / HID;
    int k = e - m * HID;
    float hx = vals_s[m] * w1[k] + b1[k];
    float g  = 0.5f * hx * (1.0f + erff(hx * 0.70710678118654752f));
    int ks = k >> 5, kk = k & 31;
    int L  = m + ((kk & 16) ? 16 : 0);
    int j  = kk & 15;
    A_lds[(ks * 32 + L) * 16 + j] = f2bf(g);
  }
  __syncthreads();

  // ---- GEMM: 48 N-tiles x 24 K-steps of v_wmma_f32_16x16x32_bf16 ----
  const int wave = tid >> 5, lane = tid & 31;
  const int nbase = wave * 6;
  // per-lane base of B fragments; ntile stride = 24*512 shorts, t stride = 512
  const unsigned short* bptr = bp + ((size_t)(nbase * 24) * 32 + lane) * 16;

#define AFRAG(t)    (*reinterpret_cast<const v16bf*>(&A_lds[((t) * 32 + lane) * 16]))
#define BFRAG(i, t) (*reinterpret_cast<const v16bf*>(bptr + (i) * (24 * 512) + (t) * 512))
#define WMMA(a_, b_, c_) \
  __builtin_amdgcn_wmma_f32_16x16x32_bf16(false, (a_), false, (b_), (short)0, (c_), false, false)

  v8f acc0 = (v8f)0.0f, acc1 = (v8f)0.0f, acc2 = (v8f)0.0f;
  v8f acc3 = (v8f)0.0f, acc4 = (v8f)0.0f, acc5 = (v8f)0.0f;

  // prologue: fragments for t = 0
  v16bf a_cur = AFRAG(0);
  v16bf f0 = BFRAG(0, 0), f1 = BFRAG(1, 0), f2 = BFRAG(2, 0);
  v16bf f3 = BFRAG(3, 0), f4 = BFRAG(4, 0), f5 = BFRAG(5, 0);

#pragma unroll
  for (int t = 0; t < 23; ++t) {
    // prefetch t+1 while computing t; fences keep the loads above the MMAs
    v16bf a_n = AFRAG(t + 1);
    v16bf n0 = BFRAG(0, t + 1), n1 = BFRAG(1, t + 1), n2 = BFRAG(2, t + 1);
    v16bf n3 = BFRAG(3, t + 1), n4 = BFRAG(4, t + 1), n5 = BFRAG(5, t + 1);
    SCHED_FENCE();
    acc0 = WMMA(a_cur, f0, acc0);
    acc1 = WMMA(a_cur, f1, acc1);
    acc2 = WMMA(a_cur, f2, acc2);
    acc3 = WMMA(a_cur, f3, acc3);
    acc4 = WMMA(a_cur, f4, acc4);
    acc5 = WMMA(a_cur, f5, acc5);
    SCHED_FENCE();
    a_cur = a_n;
    f0 = n0; f1 = n1; f2 = n2; f3 = n3; f4 = n4; f5 = n5;
  }
  // epilogue: t = 23
  acc0 = WMMA(a_cur, f0, acc0);
  acc1 = WMMA(a_cur, f1, acc1);
  acc2 = WMMA(a_cur, f2, acc2);
  acc3 = WMMA(a_cur, f3, acc3);
  acc4 = WMMA(a_cur, f4, acc4);
  acc5 = WMMA(a_cur, f5, acc5);

#undef AFRAG
#undef BFRAG
#undef WMMA

  __syncthreads();   // everyone done reading A_lds before overwriting as h_lds

  // ---- D fragments -> h_lds[16][768]; 32-bit C/D: lane<16 M=r, lane>=16 M=r+8
  {
    const int mbase = (lane >> 4) << 3;
    const int ncol0 = nbase * 16 + (lane & 15);
    v8f cc[6] = {acc0, acc1, acc2, acc3, acc4, acc5};
#pragma unroll
    for (int i = 0; i < 6; ++i) {
      int ncol = ncol0 + i * 16;
#pragma unroll
      for (int r = 0; r < 8; ++r)
        h_lds[(mbase + r) * HID + ncol] = cc[i][r];
    }
  }
  __syncthreads();

  // ---- add gathered token embeddings (L2-resident table) + b2 ----
  for (int e = tid; e < 16 * HID; e += 256) {
    int m = e / HID;
    int d = e - m * HID;
    h_lds[e] += emb[(size_t)fid_s[m] * HID + d] + b2[d];
  }
  __syncthreads();

  // ---- LayerNorm: each wave normalizes 2 tokens; NT stores for output ----
#pragma unroll
  for (int mi = 0; mi < 2; ++mi) {
    int m = wave * 2 + mi;
    float sum = 0.f, sq = 0.f;
#pragma unroll
    for (int q = 0; q < 24; ++q) {
      float v = h_lds[m * HID + lane + 32 * q];
      sum += v; sq += v * v;
    }
#pragma unroll
    for (int off = 16; off >= 1; off >>= 1) {
      sum += __shfl_xor(sum, off, 32);
      sq  += __shfl_xor(sq,  off, 32);
    }
    const float inv = 1.0f / (float)HID;
    float mu   = sum * inv;
    float var  = sq * inv - mu * mu;
    float rstd = rsqrtf(var + 1e-5f);
    bool zero_row = (empty_s != 0) && (m == 0);
    float* op = out + (size_t)(tokbase + m) * HID;
    for (int q = 0; q < 24; ++q) {
      int d = lane + 32 * q;
      float v = (h_lds[m * HID + d] - mu) * rstd * lnw[d] + lnb[d];
      if (zero_row) v = 0.0f;
      __builtin_nontemporal_store(v, op + d);
    }
  }
}

// ---------------------------------------------------------------------------
extern "C" void kernel_launch(void* const* d_in, const int* in_sizes, int n_in,
                              void* d_out, int out_size, void* d_ws, size_t ws_size,
                              hipStream_t stream) {
  const float* x   = (const float*)d_in[0];
  const float* emb = (const float*)d_in[1];
  const float* w1  = (const float*)d_in[2];
  const float* b1  = (const float*)d_in[3];
  const float* w2  = (const float*)d_in[4];
  const float* b2  = (const float*)d_in[5];
  const float* lnw = (const float*)d_in[6];
  const float* lnb = (const float*)d_in[7];

  float* out  = (float*)d_out;                                   // [B,512,768] f32
  int*   mask = (int*)(out + (size_t)BATCH * KTOK * HID);        // [B,512] i32

  char* ws = (char*)d_ws;
  unsigned short* bpack = (unsigned short*)ws;                   // 1,179,648 B
  int*   fid   = (int*)  (ws + 1179648);                         //   524,288 B
  float* vals  = (float*)(ws + 1179648 + 524288);                //   524,288 B
  int*   empty = (int*)  (ws + 1179648 + 524288 + 524288);       //     1,024 B

  hipLaunchKernelGGL(pack_w2_kernel, dim3((HID * HID + 255) / 256), dim3(256),
                     0, stream, w2, bpack);
  hipLaunchKernelGGL(topk_kernel, dim3(BATCH), dim3(256),
                     0, stream, x, fid, vals, mask, empty);
  hipLaunchKernelGGL(fused_kernel, dim3(BATCH * 32), dim3(256),
                     0, stream, emb, w1, b1, b2, lnw, lnb, bpack, fid, vals,
                     empty, out);
}
